// MambaPureTorch_36043365548405
// MI455X (gfx1250) — matrix operand order repaired and checked by reading, not compiled, not run on hardware
//
#include <hip/hip_runtime.h>
#include <hip/hip_bf16.h>
#include <math.h>

typedef __bf16 bf16_t;
typedef __attribute__((ext_vector_type(16))) __bf16 v16bf;
typedef __attribute__((ext_vector_type(8)))  __bf16 v8bf;
typedef __attribute__((ext_vector_type(8)))  float  v8f;

#define BATCH   4
#define SEQ     2048
#define DMODEL  1024
#define DINNER  2048
#define DSTATE  16
#define NROWS   (BATCH*SEQ)     /* 8192 */
#define N1      (2*DINNER)      /* 4096 */
#define NCHUNK  16
#define CHUNK   (SEQ/NCHUNK)    /* 128 */

// ---------------------------------------------------------------------------
// CDNA5 async global->LDS copy helpers (ASYNCcnt path), with fallback.
// Builtin signatures (from hipcc diagnostics): pointee types are generic
// vectors: b128 -> int vector_size(16), b64 -> int vector_size(8), b32 -> int;
// src in address_space(1), dst in address_space(3).
// ---------------------------------------------------------------------------
#if defined(__has_builtin)
#if __has_builtin(__builtin_amdgcn_global_load_async_to_lds_b128) && \
    __has_builtin(__builtin_amdgcn_global_load_async_to_lds_b64) &&  \
    __has_builtin(__builtin_amdgcn_global_load_async_to_lds_b32)
#define ASYNC_LDS 1
#endif
#endif

#if defined(ASYNC_LDS)
typedef int vs4i __attribute__((vector_size(16)));
typedef int vs2i __attribute__((vector_size(8)));
__device__ __forceinline__ void async_cp_b128(void* lds_dst, const void* gsrc) {
  __builtin_amdgcn_global_load_async_to_lds_b128(
      (__attribute__((address_space(1))) vs4i*)(void*)gsrc,
      (__attribute__((address_space(3))) vs4i*)lds_dst, 0, 0);
}
__device__ __forceinline__ void async_cp_b64(void* lds_dst, const void* gsrc) {
  __builtin_amdgcn_global_load_async_to_lds_b64(
      (__attribute__((address_space(1))) vs2i*)(void*)gsrc,
      (__attribute__((address_space(3))) vs2i*)lds_dst, 0, 0);
}
__device__ __forceinline__ void async_cp_b32(void* lds_dst, const void* gsrc) {
  __builtin_amdgcn_global_load_async_to_lds_b32(
      (__attribute__((address_space(1))) int*)(void*)gsrc,
      (__attribute__((address_space(3))) int*)lds_dst, 0, 0);
}
#else
__device__ __forceinline__ void async_cp_b128(void* lds_dst, const void* gsrc) {
  *(v8bf*)lds_dst = *(const v8bf*)gsrc;           // 16B synchronous fallback
}
__device__ __forceinline__ void async_cp_b64(void* lds_dst, const void* gsrc) {
  *(unsigned long long*)lds_dst = *(const unsigned long long*)gsrc;
}
__device__ __forceinline__ void async_cp_b32(void* lds_dst, const void* gsrc) {
  *(unsigned*)lds_dst = *(const unsigned*)gsrc;
}
#endif

#if defined(__has_builtin) && __has_builtin(__builtin_amdgcn_s_wait_asynccnt)
#define WAIT_ASYNC(n) __builtin_amdgcn_s_wait_asynccnt(n)
#else
#define WAIT_ASYNC(n) asm volatile("s_wait_asynccnt %0" :: "i"(n) : "memory")
#endif

// ---------------------------------------------------------------------------
// f32 -> bf16 conversion (bandwidth-bound, done once per operand)
// ---------------------------------------------------------------------------
__global__ void cvt_f32_to_bf16(const float* __restrict__ src,
                                bf16_t* __restrict__ dst, int n) {
  int i = blockIdx.x * blockDim.x + threadIdx.x;
  int stride = gridDim.x * blockDim.x;
  for (; i < n; i += stride) dst[i] = (bf16_t)src[i];
}

// ---------------------------------------------------------------------------
// bf16 WMMA GEMM:  C[M,N] (f32) = A[M,K] (bf16, row-major) * W[N,K]^T (bf16)
// Block: 256 threads / 8 waves, tile 32(M) x 256(N), K-step 32.
// Double-buffered LDS tiles filled with async global->LDS copies
// (5 async ops per thread per K-step: 1 x b64 for A, 4 x b128 for B).
// LDS rows padded to 40 elems (80 B = 20 banks): conflict-free ds_load_b128,
// every fragment read 16B-aligned (80 % 16 == 0).
// Fragment layouts per CDNA5 ISA §7.12.2 (wave32):
//   A lane L (m=L&15, g=L>>4): elems e -> k = (e<8 ? 8g+e : 16+8g+(e-8))
//   B lane L (n=L&15, g=L>>4): elems e -> k = (e<8 ? 8g+e : 16+8g+(e-8))  (same as A)
//   C/D vgpr c, lane L: m = c + 8g, n = L&15
// ---------------------------------------------------------------------------
#define BM 32
#define BN 256
#define BK 32
#define LSTRIDE 40                       /* padded bf16 elems per LDS row */

__global__ __launch_bounds__(256) void wmma_gemm_bf16(
    const bf16_t* __restrict__ A,
    const bf16_t* __restrict__ W,
    float* __restrict__ C,
    int M, int N, int K)
{
  __shared__ __attribute__((aligned(16))) bf16_t lsa[2][BM * LSTRIDE];
  __shared__ __attribute__((aligned(16))) bf16_t lsb[2][BN * LSTRIDE];

  const int tid  = threadIdx.x;
  const int lane = tid & 31;
  const int wave = tid >> 5;
  const int wm   = wave & 1;             // 2 waves in M
  const int wn   = wave >> 1;            // 4 waves in N
  const int g    = lane >> 4;
  const int ml   = lane & 15;

  const int tileM0 = blockIdx.y * BM;
  const int tileN0 = blockIdx.x * BN;

  // copy mapping (uniform 5 async ops per thread per K-step)
  const int ar  = tid >> 3;              // A row 0..31
  const int asc = tid & 7;               // A sub-chunk (4 bf16 = 8 B)
  const bf16_t* agsrc = A + (size_t)(tileM0 + ar) * K + asc * 4;
  const bf16_t* bgsrc = W + (size_t)(tileN0 + tid) * K;   // B row = tid

  auto issue = [&](int buf, int k0) {
    async_cp_b64(&lsa[buf][ar * LSTRIDE + asc * 4], agsrc + k0);
#pragma unroll
    for (int c = 0; c < 4; ++c)
      async_cp_b128(&lsb[buf][tid * LSTRIDE + c * 8], bgsrc + k0 + c * 8);
  };

  v8f acc[4] = {};
  const int nk = K / BK;
  issue(0, 0);

  const int arow = (wm * 16 + ml) * LSTRIDE;

  for (int it = 0; it < nk; ++it) {
    const int cur = it & 1;
    if (it + 1 < nk) {
      issue(cur ^ 1, (it + 1) * BK);
      WAIT_ASYNC(5);                      // cur buffer's 5 copies complete
    } else {
      WAIT_ASYNC(0);
    }
    __syncthreads();

    union { v16bf v; v8bf h[2]; } af;
    af.h[0] = *(const v8bf*)&lsa[cur][arow + 8 * g];        // k = 8g..8g+7
    af.h[1] = *(const v8bf*)&lsa[cur][arow + 16 + 8 * g];   // k = 16+8g..
#pragma unroll
    for (int j = 0; j < 4; ++j) {
      const int boff = (wn * 64 + j * 16 + ml) * LSTRIDE + 8 * g;
      union { v16bf v; v8bf h[2]; } bf;
      bf.h[0] = *(const v8bf*)&lsb[cur][boff];              // k = 8g..8g+7
      bf.h[1] = *(const v8bf*)&lsb[cur][boff + 16];         // k = 16+8g..
      acc[j] = __builtin_amdgcn_wmma_f32_16x16x32_bf16(
          false, af.v, false, bf.v, (short)0, acc[j], false, false);
    }
    __syncthreads();                      // before cur buffer is refilled
  }

#pragma unroll
  for (int j = 0; j < 4; ++j) {
    const int n = tileN0 + wn * 64 + j * 16 + ml;
#pragma unroll
    for (int c = 0; c < 8; ++c) {
      C[(size_t)(tileM0 + wm * 16 + c + 8 * g) * N + n] = acc[j][c];
    }
  }
}

// ---------------------------------------------------------------------------
// Depthwise causal conv (k=4) + bias + SiLU over x_inner half of xz.
// ---------------------------------------------------------------------------
__global__ void conv_silu_kernel(const float* __restrict__ xz,
                                 const float* __restrict__ conv_w,
                                 const float* __restrict__ conv_b,
                                 float* __restrict__ xc)
{
  size_t idx = (size_t)blockIdx.x * blockDim.x + threadIdx.x;
  int e = (int)(idx % DINNER);
  size_t bl = idx / DINNER;
  int l = (int)(bl % SEQ);
  size_t b = bl / SEQ;

  float acc = conv_b[e];
#pragma unroll
  for (int k = 0; k < 4; ++k) {
    int lt = l + k - 3;
    if (lt >= 0)
      acc = fmaf(conv_w[e * 4 + k], xz[((b * SEQ + lt) * (size_t)N1) + e], acc);
  }
  xc[bl * DINNER + e] = acc / (1.0f + __expf(-acc));
}

// ---------------------------------------------------------------------------
// x_ssm[bl, 0:33] = x_conv[bl, :] . W_x[s, :]    (skinny GEMM, N=33)
// One block per row; row staged into LDS via async copies.
// ---------------------------------------------------------------------------
__global__ __launch_bounds__(256) void xssm_kernel(const float* __restrict__ xc,
                                                   const float* __restrict__ Wx,
                                                   float* __restrict__ xs)
{
  __shared__ __attribute__((aligned(16))) float row[DINNER];
  const size_t bl = blockIdx.x;
  const float* src = xc + bl * DINNER;
  for (int i = threadIdx.x; i < DINNER / 4; i += 256)      // 16B chunks
    async_cp_b128(&row[i * 4], &src[i * 4]);
  WAIT_ASYNC(0);
  __syncthreads();

  const int lane = threadIdx.x & 31;
  const int wave = threadIdx.x >> 5;
  for (int s = wave; s < 2 * DSTATE + 1; s += 8) {
    const float* wrow = Wx + (size_t)s * DINNER;
    float p = 0.f;
    for (int k = lane; k < DINNER; k += 32) p = fmaf(row[k], wrow[k], p);
    for (int off = 16; off > 0; off >>= 1) p += __shfl_down(p, off, 32);
    if (lane == 0) xs[bl * (2 * DSTATE + 1) + s] = p;
  }
}

// ---------------------------------------------------------------------------
// Chunked linear-recurrence scan. Phase A: per-chunk local scan from h=0;
// store end state and running product of dA.
// ---------------------------------------------------------------------------
__global__ __launch_bounds__(256) void scan_phaseA(
    const float* __restrict__ xs, const float* __restrict__ xc,
    const float* __restrict__ A_log, const float* __restrict__ W_dt,
    const float* __restrict__ b_dt,
    float* __restrict__ hend, float* __restrict__ pprod)
{
  __shared__ __attribute__((aligned(16))) float tile[CHUNK * 33];
  const int blk  = blockIdx.x;                     // B * NCHUNK * (DINNER/256)
  const int egrp = blk % (DINNER / 256);
  const int c    = (blk / (DINNER / 256)) % NCHUNK;
  const int b    = blk / (DINNER / 256) / NCHUNK;
  const int e    = egrp * 256 + threadIdx.x;
  const int t0   = c * CHUNK;

  const float* src = xs + ((size_t)b * SEQ + t0) * 33;     // only 4B-aligned
  for (int i = threadIdx.x; i < CHUNK * 33; i += 256)
    async_cp_b32(&tile[i], &src[i]);
  WAIT_ASYNC(0);
  __syncthreads();

  float Ar[DSTATE];
#pragma unroll
  for (int s = 0; s < DSTATE; ++s) Ar[s] = -__expf(A_log[e * DSTATE + s]);
  const float wdt = W_dt[e], bdt = b_dt[e];

  float h[DSTATE], P[DSTATE];
#pragma unroll
  for (int s = 0; s < DSTATE; ++s) { h[s] = 0.f; P[s] = 1.f; }

  for (int tt = 0; tt < CHUNK; ++tt) {
    const float* ts = tile + tt * 33;
    float u  = fmaf(ts[32], wdt, bdt);
    float dt = (u > 20.f) ? u : __logf(1.f + __expf(u));
    float xt = xc[((size_t)b * SEQ + t0 + tt) * DINNER + e];
    float dx = dt * xt;
#pragma unroll
    for (int s = 0; s < DSTATE; ++s) {
      float dA = __expf(Ar[s] * dt);
      h[s] = fmaf(dA, h[s], dx * ts[s]);
      P[s] *= dA;
    }
  }

  size_t base = (((size_t)b * NCHUNK + c) * DINNER + e) * DSTATE;
#pragma unroll
  for (int s = 0; s < DSTATE; ++s) { hend[base + s] = h[s]; pprod[base + s] = P[s]; }
}

// Phase B: chain chunk boundary states sequentially (16 steps x 16 states).
__global__ void scan_phaseB(const float* __restrict__ hend,
                            const float* __restrict__ pprod,
                            float* __restrict__ hin)
{
  int idx = blockIdx.x * blockDim.x + threadIdx.x;   // B * DINNER
  int e = idx % DINNER;
  int b = idx / DINNER;
  float h[DSTATE];
#pragma unroll
  for (int s = 0; s < DSTATE; ++s) h[s] = 0.f;
  for (int c = 0; c < NCHUNK; ++c) {
    size_t base = (((size_t)b * NCHUNK + c) * DINNER + e) * DSTATE;
#pragma unroll
    for (int s = 0; s < DSTATE; ++s) {
      hin[base + s] = h[s];
      h[s] = fmaf(pprod[base + s], h[s], hend[base + s]);
    }
  }
}

// Phase C: re-scan each chunk from true initial state; fuse y = C.h + D*x,
// SiLU(z) gating, and bf16 cast for the output GEMM.
__global__ __launch_bounds__(256) void scan_phaseC(
    const float* __restrict__ xs, const float* __restrict__ xc,
    const float* __restrict__ xz,
    const float* __restrict__ A_log, const float* __restrict__ W_dt,
    const float* __restrict__ b_dt, const float* __restrict__ D_param,
    const float* __restrict__ hin,
    bf16_t* __restrict__ yb)
{
  __shared__ __attribute__((aligned(16))) float tile[CHUNK * 33];
  const int blk  = blockIdx.x;
  const int egrp = blk % (DINNER / 256);
  const int c    = (blk / (DINNER / 256)) % NCHUNK;
  const int b    = blk / (DINNER / 256) / NCHUNK;
  const int e    = egrp * 256 + threadIdx.x;
  const int t0   = c * CHUNK;

  const float* src = xs + ((size_t)b * SEQ + t0) * 33;
  for (int i = threadIdx.x; i < CHUNK * 33; i += 256)
    async_cp_b32(&tile[i], &src[i]);
  WAIT_ASYNC(0);
  __syncthreads();

  float Ar[DSTATE];
#pragma unroll
  for (int s = 0; s < DSTATE; ++s) Ar[s] = -__expf(A_log[e * DSTATE + s]);
  const float wdt = W_dt[e], bdt = b_dt[e], Dp = D_param[e];

  float h[DSTATE];
  size_t base = (((size_t)b * NCHUNK + c) * DINNER + e) * DSTATE;
#pragma unroll
  for (int s = 0; s < DSTATE; ++s) h[s] = hin[base + s];

  for (int tt = 0; tt < CHUNK; ++tt) {
    const int t = t0 + tt;
    const float* ts = tile + tt * 33;
    float u  = fmaf(ts[32], wdt, bdt);
    float dt = (u > 20.f) ? u : __logf(1.f + __expf(u));
    float xt = xc[((size_t)b * SEQ + t) * DINNER + e];
    float dx = dt * xt;
    float y = 0.f;
#pragma unroll
    for (int s = 0; s < DSTATE; ++s) {
      float dA = __expf(Ar[s] * dt);
      h[s] = fmaf(dA, h[s], dx * ts[s]);
      y = fmaf(h[s], ts[DSTATE + s], y);
    }
    y = fmaf(xt, Dp, y);
    float zt = xz[((size_t)b * SEQ + t) * N1 + DINNER + e];
    y *= zt / (1.0f + __expf(-zt));
    yb[((size_t)b * SEQ + t) * DINNER + e] = (bf16_t)y;
  }
}

// ---------------------------------------------------------------------------
extern "C" void kernel_launch(void* const* d_in, const int* in_sizes, int n_in,
                              void* d_out, int out_size, void* d_ws, size_t ws_size,
                              hipStream_t stream)
{
  (void)in_sizes; (void)n_in; (void)out_size; (void)ws_size;
  const float* x      = (const float*)d_in[0];
  const float* W_in   = (const float*)d_in[1];
  const float* conv_w = (const float*)d_in[2];
  const float* conv_b = (const float*)d_in[3];
  const float* W_x    = (const float*)d_in[4];
  const float* W_dt   = (const float*)d_in[5];
  const float* b_dt   = (const float*)d_in[6];
  const float* A_log  = (const float*)d_in[7];
  const float* D_par  = (const float*)d_in[8];
  const float* W_out  = (const float*)d_in[9];

  char* ws = (char*)d_ws;
  size_t off = 0;
  auto alloc = [&](size_t bytes) -> void* {
    void* p = ws + off;
    off = (off + bytes + 255) & ~(size_t)255;
    return p;
  };
  bf16_t* xb   = (bf16_t*)alloc((size_t)NROWS * DMODEL * 2);
  bf16_t* wib  = (bf16_t*)alloc((size_t)N1 * DMODEL * 2);
  bf16_t* wob  = (bf16_t*)alloc((size_t)DMODEL * DINNER * 2);
  float*  xzb  = (float*) alloc((size_t)NROWS * N1 * 4);
  float*  xcb  = (float*) alloc((size_t)NROWS * DINNER * 4);
  float*  xsb  = (float*) alloc((size_t)NROWS * 33 * 4);
  float*  hend = (float*) alloc((size_t)BATCH * NCHUNK * DINNER * DSTATE * 4);
  float*  ppr  = (float*) alloc((size_t)BATCH * NCHUNK * DINNER * DSTATE * 4);
  float*  hin  = (float*) alloc((size_t)BATCH * NCHUNK * DINNER * DSTATE * 4);
  bf16_t* yb   = (bf16_t*)alloc((size_t)NROWS * DINNER * 2);

  // operand conversions to bf16
  {
    int n0 = NROWS * DMODEL, n1 = N1 * DMODEL, n2 = DMODEL * DINNER;
    cvt_f32_to_bf16<<<(n0 + 255) / 256, 256, 0, stream>>>(x, xb, n0);
    cvt_f32_to_bf16<<<(n1 + 255) / 256, 256, 0, stream>>>(W_in, wib, n1);
    cvt_f32_to_bf16<<<(n2 + 255) / 256, 256, 0, stream>>>(W_out, wob, n2);
  }

  // GEMM1: xz = x . W_in^T   (8192 x 4096 x 1024)
  {
    dim3 grid(N1 / BN, NROWS / BM);
    wmma_gemm_bf16<<<grid, 256, 0, stream>>>(xb, wib, xzb, NROWS, N1, DMODEL);
  }

  // depthwise conv + SiLU
  conv_silu_kernel<<<(NROWS * DINNER) / 256, 256, 0, stream>>>(xzb, conv_w, conv_b, xcb);

  // x_ssm = x_conv . W_x^T   (N = 33)
  xssm_kernel<<<NROWS, 256, 0, stream>>>(xcb, W_x, xsb);

  // chunked selective scan (+ D skip + SiLU(z) gate, bf16 output)
  const int scan_blocks = BATCH * NCHUNK * (DINNER / 256);
  scan_phaseA<<<scan_blocks, 256, 0, stream>>>(xsb, xcb, A_log, W_dt, b_dt, hend, ppr);
  scan_phaseB<<<(BATCH * DINNER) / 256, 256, 0, stream>>>(hend, ppr, hin);
  scan_phaseC<<<scan_blocks, 256, 0, stream>>>(xsb, xcb, xzb, A_log, W_dt, b_dt,
                                               D_par, hin, yb);

  // GEMM2: out = y . W_out^T   (8192 x 1024 x 2048)
  {
    dim3 grid(DMODEL / BN, NROWS / BM);
    wmma_gemm_bf16<<<grid, 256, 0, stream>>>(yb, wob, (float*)d_out, NROWS, DMODEL, DINNER);
  }
}
